// ContentSelectionCell_38792144618058
// MI455X (gfx1250) — matrix operand order repaired and checked by reading, not compiled, not run on hardware
//
#include <hip/hip_runtime.h>
#include <cstdint>

// Problem constants (from reference): B=64, T=2048, D=H=512
#define B_ 64
#define T_ 2048
#define D_ 512
#define H_ 512
#define S_ 8                 // splits of T across workgroups
#define TSPLIT (T_ / S_)     // 256 timesteps per workgroup
#define TTILE 64             // timesteps per LDS tile (64*512*4 = 128KB)
#define NTILE (TSPLIT / TTILE)

typedef __attribute__((ext_vector_type(16))) _Float16 v16h;
typedef __attribute__((ext_vector_type(8)))  _Float16 v8h;
typedef __attribute__((ext_vector_type(8)))  float    v8f;
typedef __attribute__((ext_vector_type(4)))  unsigned int u32x4;
typedef __attribute__((ext_vector_type(8)))  int i32x8;
typedef __attribute__((ext_vector_type(4)))  int i32x4;

// ---------------------------------------------------------------------------
// TDM: async DMA of a [TTILE x D_] f32 tile (row stride D_) from global -> LDS
// D# per CDNA5 ISA (2D tile, groups 2/3 zero).
// This toolchain exposes the 6-arg builtin:
//   (uint32x4 g0, int32x8 g1, int32x4 g2, int32x4 g3, int32x8 pad, i32 cpol)
// ---------------------------------------------------------------------------
__device__ __forceinline__ void tdm_load_tile(const float* gptr, unsigned lds_byte_off) {
    unsigned long long ga = (unsigned long long)(uintptr_t)gptr;
    u32x4 g0;
    g0[0] = 1u;                                        // count=1 (valid user D#)
    g0[1] = lds_byte_off;                              // lds_addr
    g0[2] = (unsigned)(ga & 0xFFFFFFFFu);              // global_addr[31:0]
    g0[3] = (unsigned)((ga >> 32) & 0x01FFFFFFu)       // global_addr[56:32]
          | 0x80000000u;                               // type=2 ("image")
    i32x8 g1;
    g1[0] = 0x00020000;                                // data_size=2 (4 bytes)
    g1[1] = (int)(512u << 16);                         // tensor_dim0 = 512 (bits 63:48)
    g1[2] = (int)((unsigned)TTILE << 16);              // tensor_dim1 = 64 (bits 95:80)
    g1[3] = (int)(512u << 16);                         // tile_dim0 = 512 (bits 127:112)
    g1[4] = TTILE;                                     // tile_dim1 = 64 (bits 143:128)
    g1[5] = 512;                                       // tensor_dim0_stride = 512
    g1[6] = 0;
    g1[7] = 0;
    i32x4 z4 = {0, 0, 0, 0};
    i32x8 z8 = {0, 0, 0, 0, 0, 0, 0, 0};
    __builtin_amdgcn_tensor_load_to_lds(g0, g1, z4, z4, z8, 0);
}

// ---------------------------------------------------------------------------
// Kernel 1: per-(split, batch) fused scores + online softmax + context.
// Each WG: 256 threads (8 waves), double-buffered 128KB LDS tiles fed by TDM.
// Reads enc_outs from HBM exactly once.
// ---------------------------------------------------------------------------
__global__ void attn_partial_kernel(const float* __restrict__ x,
                                    const float* __restrict__ enc,
                                    const int*   __restrict__ step_ptr,
                                    float* __restrict__ pctx,
                                    float* __restrict__ pm,
                                    float* __restrict__ pl) {
    extern __shared__ __align__(16) char smem[];
    float* tile = (float*)smem;                 // [2][TTILE*D_]  (at LDS offset 0)
    float* xs   = tile + 2 * TTILE * D_;        // [D_]
    float* sc   = xs + D_;                      // [TTILE] raw scores
    float* pw   = sc + TTILE;                   // [TTILE] exp weights

    const int s    = blockIdx.x;
    const int b    = blockIdx.y;
    const int tid  = threadIdx.x;
    const int w    = tid >> 5;
    const int lane = tid & 31;
    const int step = *step_ptr;
    const int tbase = s * TSPLIT;
    const float* encb = enc + (size_t)b * T_ * D_;

    // Prologue: kick off DMA of tile 0, stage x row in LDS.
    if (w == 0) tdm_load_tile(encb + (size_t)tbase * D_, 0u);
    xs[tid]       = x[b * D_ + tid];
    xs[tid + 256] = x[b * D_ + tid + 256];

    float m_run = -1e30f, l_run = 0.0f, ctx0 = 0.0f, ctx1 = 0.0f;
    const int d0 = tid, d1 = tid + 256;

    for (int t = 0; t < NTILE; ++t) {
        const int buf = t & 1;
        if (w == 0) {
            if (t + 1 < NTILE) {
                tdm_load_tile(encb + (size_t)(tbase + (t + 1) * TTILE) * D_,
                              (unsigned)((buf ^ 1) * TTILE * D_ * 4));
                __builtin_amdgcn_s_wait_tensorcnt(1);   // tile t landed, t+1 in flight
            } else {
                __builtin_amdgcn_s_wait_tensorcnt(0);
            }
        }
        __syncthreads();   // tile[buf] (and xs on t==0) visible to all waves

        const float* tb = tile + buf * TTILE * D_;

        // Scores: wave w handles 8 timesteps; 32-lane dot over D=512 from LDS.
        for (int i = 0; i < 8; ++i) {
            const int tl = w * 8 + i;
            const float* row = tb + tl * D_;
            float acc = 0.0f;
#pragma unroll
            for (int k = 0; k < D_ / 32; ++k)
                acc = fmaf(xs[lane + 32 * k], row[lane + 32 * k], acc);
#pragma unroll
            for (int off = 16; off > 0; off >>= 1)
                acc += __shfl_xor(acc, off, 32);
            if (lane == 0) {
                const int tg = tbase + t * TTILE + tl;
                sc[tl] = (tg == step) ? -1e9f : acc;
            }
        }
        __syncthreads();

        // Online softmax update (redundant uniform math across threads).
        float mt = -1e30f;
        for (int i = 0; i < TTILE; ++i) mt = fmaxf(mt, sc[i]);
        const float m_new   = fmaxf(m_run, mt);
        const float rescale = __expf(m_run - m_new);
        if (tid < TTILE) pw[tid] = __expf(sc[tid] - m_new);
        __syncthreads();

        float lt = 0.0f;
        for (int i = 0; i < TTILE; ++i) lt += pw[i];
        l_run = l_run * rescale + lt;
        m_run = m_new;
        ctx0 *= rescale;
        ctx1 *= rescale;
        for (int i = 0; i < TTILE; ++i) {
            const float pv = pw[i];
            ctx0 = fmaf(pv, tb[i * D_ + d0], ctx0);
            ctx1 = fmaf(pv, tb[i * D_ + d1], ctx1);
        }
        __syncthreads();   // done with tile[buf] / sc / pw
    }

    const size_t pi = (size_t)s * B_ + b;
    pctx[pi * D_ + d0] = ctx0;
    pctx[pi * D_ + d1] = ctx1;
    if (tid == 0) { pm[pi] = m_run; pl[pi] = l_run; }
}

// ---------------------------------------------------------------------------
// Kernel 2: merge split partials -> context; emit f16 Acat = [x | context].
// ---------------------------------------------------------------------------
__global__ void combine_kernel(const float* __restrict__ x,
                               const float* __restrict__ pctx,
                               const float* __restrict__ pm,
                               const float* __restrict__ pl,
                               _Float16* __restrict__ Acat) {
    const int b = blockIdx.x, tid = threadIdx.x;
    const int d0 = tid, d1 = tid + 256;
    float M = -1e30f;
    for (int s = 0; s < S_; ++s) M = fmaxf(M, pm[(size_t)s * B_ + b]);
    float L = 0.0f, c0 = 0.0f, c1 = 0.0f;
    for (int s = 0; s < S_; ++s) {
        const size_t pi  = (size_t)s * B_ + b;
        const float  wgt = __expf(pm[pi] - M);
        L  += pl[pi] * wgt;
        c0 += pctx[pi * D_ + d0] * wgt;
        c1 += pctx[pi * D_ + d1] * wgt;
    }
    const float inv = 1.0f / L;
    _Float16* row = Acat + (size_t)b * (2 * D_);
    row[d0]       = (_Float16)x[b * D_ + d0];
    row[d1]       = (_Float16)x[b * D_ + d1];
    row[D_ + d0]  = (_Float16)(c0 * inv);
    row[D_ + d1]  = (_Float16)(c1 * inv);
}

// ---------------------------------------------------------------------------
// Kernel 3: W (f32 [1024x512]) -> f16
// ---------------------------------------------------------------------------
__global__ void wconv_kernel(const float* __restrict__ W, _Float16* __restrict__ Wh) {
    const int i = blockIdx.x * blockDim.x + threadIdx.x;
    if (i < 2 * D_ * H_) Wh[i] = (_Float16)W[i];
}

// ---------------------------------------------------------------------------
// Kernel 4: gate GEMM [64x1024]@[1024x512] via V_WMMA_F32_16X16X32_F16,
// fused bias + sigmoid + (* x). One wave per 16x16 output tile.
// A layout (ISA 16-bit A 16x32): lane<16 -> row=lane, K {0..7,16..23};
//                                lane>=16 -> row=lane-16, K {8..15,24..31}.
// B layout: N = lane%16, K same as A (lane<16: {0..7,16..23}; lane>=16: {8..15,24..31}).
// C layout: VGPR v -> M = v (lanes 0-15) / v+8 (lanes 16-31), N = lane%16.
// ---------------------------------------------------------------------------
__global__ void gemm_gate_kernel(const _Float16* __restrict__ A,
                                 const _Float16* __restrict__ Bw,
                                 const float* __restrict__ bias,
                                 const float* __restrict__ x,
                                 float* __restrict__ out) {
    const int tid  = threadIdx.x;
    const int wv   = tid >> 5;
    const int lane = tid & 31;
    const int tileid = blockIdx.x * 4 + wv;     // 0..127 (4 x 32 tiles)
    const int tm = tileid >> 5;                 // 0..3   (M tiles of 16)
    const int tn = tileid & 31;                 // 0..31  (N tiles of 16)
    const int mrow = tm * 16 + (lane & 15);
    const int kofs = (lane < 16) ? 0 : 8;

    v8f c = {};
    for (int k0 = 0; k0 < 2 * D_; k0 += 32) {
        const _Float16* ar = A + (size_t)mrow * (2 * D_) + k0 + kofs;
        const v8h lo = *(const v8h*)ar;          // K = k0+kofs .. +7
        const v8h hi = *(const v8h*)(ar + 16);   // K = k0+kofs+16 .. +23
        v16h a;
#pragma unroll
        for (int i = 0; i < 8; ++i) { a[i] = lo[i]; a[i + 8] = hi[i]; }
        v16h bm;
#pragma unroll
        for (int i = 0; i < 8; ++i) {
            bm[i]     = Bw[(size_t)(k0 + kofs + i) * H_ + tn * 16 + (lane & 15)];
            bm[i + 8] = Bw[(size_t)(k0 + kofs + 16 + i) * H_ + tn * 16 + (lane & 15)];
        }
        c = __builtin_amdgcn_wmma_f32_16x16x32_f16(
                /*neg_a=*/false, a, /*neg_b=*/false, bm,
                /*c_mod=*/(short)0, c, /*reuse_a=*/false, /*reuse_b=*/false);
    }

    const int col  = tn * 16 + (lane & 15);
    const int rsel = (lane < 16) ? 0 : 8;
#pragma unroll
    for (int v = 0; v < 8; ++v) {
        const int r = tm * 16 + v + rsel;
        const float g   = c[v] + bias[col];
        const float sig = 1.0f / (1.0f + __expf(-g));
        out[(size_t)r * H_ + col] = sig * x[(size_t)r * D_ + col];
    }
}

// ---------------------------------------------------------------------------
extern "C" void kernel_launch(void* const* d_in, const int* in_sizes, int n_in,
                              void* d_out, int out_size, void* d_ws, size_t ws_size,
                              hipStream_t stream) {
    const float* x    = (const float*)d_in[0];   // [64, 512]
    const float* enc  = (const float*)d_in[1];   // [64, 2048, 512]
    const float* W    = (const float*)d_in[2];   // [1024, 512]
    const float* bias = (const float*)d_in[3];   // [512]
    const int*   step = (const int*)d_in[4];     // scalar
    float*       out  = (float*)d_out;           // [64, 512]

    // Workspace layout (~2.2 MB)
    char* ws = (char*)d_ws;
    float* pctx = (float*)ws;                                        // S*B*D f32 = 1 MiB
    size_t off  = (size_t)S_ * B_ * D_ * sizeof(float);
    float* pm   = (float*)(ws + off);  off += (size_t)S_ * B_ * sizeof(float);
    float* pl   = (float*)(ws + off);  off += (size_t)S_ * B_ * sizeof(float);
    _Float16* Acat = (_Float16*)(ws + off);  off += (size_t)B_ * 2 * D_ * sizeof(_Float16);
    _Float16* Wh   = (_Float16*)(ws + off);

    const size_t shmem = (size_t)(2 * TTILE * D_ + D_ + 2 * TTILE) * sizeof(float); // ~258.5 KB
    (void)hipFuncSetAttribute((const void*)attn_partial_kernel,
                              hipFuncAttributeMaxDynamicSharedMemorySize, (int)shmem);

    wconv_kernel<<<dim3((2 * D_ * H_ + 511) / 512), dim3(512), 0, stream>>>(W, Wh);
    attn_partial_kernel<<<dim3(S_, B_), dim3(256), shmem, stream>>>(x, enc, step, pctx, pm, pl);
    combine_kernel<<<dim3(B_), dim3(256), 0, stream>>>(x, pctx, pm, pl, Acat);
    gemm_gate_kernel<<<dim3(32), dim3(128), 0, stream>>>(Acat, Wh, bias, x, out);
}
